// DeformableTransformerDecoderLayer_82884278879248
// MI455X (gfx1250) — compile-verified, matrix operand order checked by reading, not run on hardware
//
#include <hip/hip_runtime.h>
#include <hip/hip_bf16.h>

typedef __attribute__((ext_vector_type(16))) _Float16 v16h;
typedef __attribute__((ext_vector_type(8)))  _Float16 v8h;
typedef __attribute__((ext_vector_type(8)))  float    v8f;

#define B_   8
#define C_   256
#define H_   8
#define DH_  32
#define Q_   900
#define QP_  928      // Q padded to multiple of 32 (GEMM M tiles + K dim)
#define KP_  928
#define M_   (B_*Q_)  // 7200 tokens total (multiple of 32)

// ---------------------------------------------------------------------------
// WMMA fragment load: row-major f16 matrix, 16x32 fragment (A layout; B uses
// the mirrored layout with W stored as [N,K] row-major so the same loader
// works for both operands). Two contiguous 16B loads per lane.
// Layout per CDNA5 ISA 7.12.2 (16-bit A 16x32):
//   lanes 0-15 : M=lane,    halves 0-7 -> K=k0+0..7,  halves 8-15 -> K=k0+16..23
//   lanes 16-31: M=lane-16, halves 0-7 -> K=k0+8..15, halves 8-15 -> K=k0+24..31
// ---------------------------------------------------------------------------
__device__ __forceinline__ v16h load_frag(const _Float16* base, long row, long ld,
                                          int k0, int lane) {
  int kg = (lane & 16) ? 8 : 0;
  const _Float16* p = base + row * ld + (long)(k0 + kg);
  v8h lo = *(const v8h*)(p);
  v8h hi = *(const v8h*)(p + 16);
  v16h r;
#pragma unroll
  for (int i = 0; i < 8; ++i) { r[i] = lo[i]; r[i + 8] = hi[i]; }
  return r;
}

struct GemmP {
  const _Float16* A; long lda;  long strideA;   // A[M,K] row-major (per batch)
  const _Float16* W; long ldw;  long strideW;   // W[N,K] row-major (per batch)
  const float*    bias;                          // indexed by n, may be null
  void*           out; long ldo; long strideO;
  int M, N, K, Mtiles, NG;                       // Mtiles = M/32, NG = ceil(N/64)
};

// MODE: 0=f32 out, 1=f32 sigmoid, 2=f16 out, 3=f16 relu,
//       4=f16 scatter [b,h,QP_,32], 5=f16 scatter transposed [b,h,32,KP_],
//       6=f16 batched scores (ld=KP_), 7=f16 batched scatter back to tokens
// Per-wave tile: 32(M) x 64(N) -> 8 WMMAs per K-step, each B fragment reused
// by two A fragments (2x arithmetic intensity vs 16x64 tiles).
template <int MODE>
__global__ __launch_bounds__(256) void gemm_wmma(GemmP p) {
  int wid  = (int)((blockIdx.x * blockDim.x + threadIdx.x) >> 5);
  int lane = (int)(threadIdx.x & 31);
  int total = p.Mtiles * p.NG;
  if (wid >= total) return;
  int mi = wid % p.Mtiles;
  int ng = wid / p.Mtiles;
  int Mb = mi * 32, Nb = ng * 64;
  int bh = (int)blockIdx.y;

  const _Float16* A = p.A + (long)bh * p.strideA;
  const _Float16* W = p.W + (long)bh * p.strideW;

  v8f acc0[4], acc1[4];
#pragma unroll
  for (int j = 0; j < 4; ++j)
#pragma unroll
    for (int i = 0; i < 8; ++i) { acc0[j][i] = 0.0f; acc1[j][i] = 0.0f; }

  int r = lane & 15;
  long Arow0 = Mb + r;
  long Arow1 = Mb + 16 + r;

  for (int k0 = 0; k0 < p.K; k0 += 32) {
    v16h a0 = load_frag(A, Arow0, p.lda, k0, lane);
    v16h a1 = load_frag(A, Arow1, p.lda, k0, lane);
#pragma unroll
    for (int j = 0; j < 4; ++j) {
      int n  = Nb + j * 16 + r;
      long nr = (n < p.N) ? n : (p.N - 1);   // clamp for narrow-N batches
      v16h b = load_frag(W, nr, p.ldw, k0, lane);
      acc0[j] = __builtin_amdgcn_wmma_f32_16x16x32_f16(
          false, a0, false, b, (short)0, acc0[j], false, false);
      acc1[j] = __builtin_amdgcn_wmma_f32_16x16x32_f16(
          false, a1, false, b, (short)0, acc1[j], false, false);
    }
  }

  // C/D layout: VGPR r2 -> M = Mb + r2 + (lane>=16 ? 8 : 0), N = Nb + j*16 + (lane&15)
  int nl   = lane & 15;
  int moff = (lane & 16) ? 8 : 0;
#pragma unroll
  for (int half = 0; half < 2; ++half) {
    int MbH = Mb + half * 16;
#pragma unroll
    for (int j = 0; j < 4; ++j) {
      int n = Nb + j * 16 + nl;
      if (n >= p.N) continue;
      float bv = p.bias ? p.bias[n] : 0.0f;
#pragma unroll
      for (int r2 = 0; r2 < 8; ++r2) {
        int m   = MbH + r2 + moff;
        float v = (half == 0 ? acc0[j][r2] : acc1[j][r2]) + bv;
        if constexpr (MODE == 0) {
          ((float*)p.out)[(long)m * p.ldo + n] = v;
        } else if constexpr (MODE == 1) {
          ((float*)p.out)[(long)m * p.ldo + n] = 1.0f / (1.0f + __expf(-v));
        } else if constexpr (MODE == 2) {
          ((_Float16*)p.out)[(long)m * p.ldo + n] = (_Float16)v;
        } else if constexpr (MODE == 3) {
          ((_Float16*)p.out)[(long)m * p.ldo + n] = (_Float16)fmaxf(v, 0.0f);
        } else if constexpr (MODE == 4) {
          int b = m / Q_, q = m % Q_;
          int h = n >> 5, d = n & 31;
          ((_Float16*)p.out)[((long)(b * H_ + h) * QP_ + q) * DH_ + d] = (_Float16)v;
        } else if constexpr (MODE == 5) {
          int b = m / Q_, q = m % Q_;
          int h = n >> 5, d = n & 31;
          ((_Float16*)p.out)[((long)(b * H_ + h) * DH_ + d) * KP_ + q] = (_Float16)v;
        } else if constexpr (MODE == 6) {
          ((_Float16*)p.out)[(long)bh * p.strideO + (long)m * p.ldo + n] = (_Float16)v;
        } else if constexpr (MODE == 7) {
          if (m < Q_) {
            int b = bh >> 3, h = bh & 7;
            ((_Float16*)p.out)[((long)b * Q_ + m) * C_ + h * DH_ + n] = (_Float16)v;
          }
        }
      }
    }
  }
}

// ---------------------------------------------------------------------------
// Elementwise / reduction helper kernels
// ---------------------------------------------------------------------------
__global__ void f32_to_f16_kernel(const float* __restrict__ s, _Float16* __restrict__ d, int n) {
  int i = (int)(blockIdx.x * blockDim.x + threadIdx.x);
  if (i < n) d[i] = (_Float16)s[i];
}

// tokens: [B,C,1,Q] -> row-major [B*Q, C]
__global__ void pack_tokens_kernel(const float* __restrict__ tgt, const float* __restrict__ qp,
                                   _Float16* __restrict__ Xqk, _Float16* __restrict__ Xv,
                                   float* __restrict__ T0, float* __restrict__ QPt) {
  long idx = (long)blockIdx.x * blockDim.x + threadIdx.x;
  if (idx >= (long)M_ * C_) return;
  int c = (int)(idx & 255);
  long row = idx >> 8;
  int b = (int)(row / Q_), q = (int)(row % Q_);
  float t = tgt[((long)b * C_ + c) * Q_ + q];
  float p = qp [((long)b * C_ + c) * Q_ + q];
  Xqk[idx] = (_Float16)(t + p);
  Xv [idx] = (_Float16)t;
  T0 [idx] = t;
  QPt[idx] = p;
}

// one feature level: [B,C,HW] -> row-major [B*HW, C] f16
__global__ void pack_values_kernel(const float* __restrict__ src, _Float16* __restrict__ dst,
                                   int HW, long total) {
  long idx = (long)blockIdx.x * blockDim.x + threadIdx.x;
  if (idx >= total) return;
  int c = (int)(idx & 255);
  long row = idx >> 8;
  int b = (int)(row / HW);
  long pix = row % HW;
  dst[idx] = (_Float16)src[((long)b * C_ + c) * HW + pix];
}

// masked softmax over scores rows: S is [64, QP_, KP_] f16, valid region 900x900
__global__ void softmax_kernel(_Float16* __restrict__ S) {
  long row = blockIdx.x;              // 0 .. 64*QP_-1
  int  m   = (int)(row % QP_);
  _Float16* s = S + row * KP_;
  int tid = (int)threadIdx.x;
  if (m >= Q_) {                      // padded rows -> all-zero probabilities
    for (int j = tid; j < KP_; j += 256) s[j] = (_Float16)0.0f;
    return;
  }
  const float scale = 0.17677669529663688f;   // 1/sqrt(32)
  __shared__ float red[256];
  float t[4];
  int cnt = 0;
  float mx = -3.0e38f;
  for (int j = tid; j < Q_; j += 256) { float v = (float)s[j] * scale; t[cnt++] = v; mx = fmaxf(mx, v); }
  red[tid] = mx; __syncthreads();
  for (int o = 128; o > 0; o >>= 1) { if (tid < o) red[tid] = fmaxf(red[tid], red[tid + o]); __syncthreads(); }
  mx = red[0]; __syncthreads();
  float sum = 0.0f;
  for (int i = 0; i < cnt; ++i) { t[i] = __expf(t[i] - mx); sum += t[i]; }
  red[tid] = sum; __syncthreads();
  for (int o = 128; o > 0; o >>= 1) { if (tid < o) red[tid] += red[tid + o]; __syncthreads(); }
  float inv = 1.0f / red[0];
  cnt = 0;
  for (int j = tid; j < Q_; j += 256) s[j] = (_Float16)(t[cnt++] * inv);
  for (int j = Q_ + tid; j < KP_; j += 256) s[j] = (_Float16)0.0f;  // zero pad K region
}

// LayerNorm over C=256 (one block per token, one thread per channel).
// v = x + r; y = LN(v)*g+b. Optional: y32, y16=f16(y+addh), transposed f32 out.
__global__ void ln_kernel(const float* __restrict__ x, const float* __restrict__ r,
                          const float* __restrict__ g, const float* __restrict__ bb,
                          float* y32, _Float16* y16, const float* addh, float* yT) {
  int row = (int)blockIdx.x;
  int c   = (int)threadIdx.x;
  long idx = (long)row * C_ + c;
  float v = x[idx] + r[idx];
  __shared__ float red[256];
  __shared__ float s_mean, s_var;
  red[c] = v; __syncthreads();
  for (int o = 128; o > 0; o >>= 1) { if (c < o) red[c] += red[c + o]; __syncthreads(); }
  if (c == 0) s_mean = red[0] * (1.0f / 256.0f);
  __syncthreads();
  float d = v - s_mean;
  red[c] = d * d; __syncthreads();
  for (int o = 128; o > 0; o >>= 1) { if (c < o) red[c] += red[c + o]; __syncthreads(); }
  if (c == 0) s_var = red[0] * (1.0f / 256.0f);
  __syncthreads();
  float y = d * rsqrtf(s_var + 1e-5f) * g[c] + bb[c];
  if (y32) y32[idx] = y;
  if (y16) y16[idx] = (_Float16)(y + (addh ? addh[idx] : 0.0f));
  if (yT) {
    int b = row / Q_, q = row % Q_;
    yT[((long)b * C_ + c) * Q_ + q] = y;
  }
}

__device__ __forceinline__ float sample_corner(const _Float16* __restrict__ vp, long base,
                                               int vw, int vh, int xi, int yi, float cw, int col) {
  if (xi < 0 || yi < 0 || xi >= vw || yi >= vh) return 0.0f;
  return cw * (float)vp[(base + (long)yi * vw + xi) * C_ + col];
}

// one wave per (token,head); lane = channel within head
__global__ __launch_bounds__(256) void deform_sample_kernel(
    const float* __restrict__ off, const float* __restrict__ aw,
    const float* __restrict__ ref, const _Float16* __restrict__ vproj,
    _Float16* __restrict__ outTok) {
  int wid  = (int)((blockIdx.x * blockDim.x + threadIdx.x) >> 5);
  int lane = (int)(threadIdx.x & 31);
  int h  = wid & 7;
  int bq = wid >> 3;
  if (bq >= M_) return;
  int b = bq / Q_, q = bq % Q_;
  const int  vhA[4] = {92, 46, 23, 12};
  const int  vwA[4] = {160, 80, 40, 20};
  const int  hwA[4] = {14720, 3680, 920, 240};
  const long baseRow[4] = {0, (long)B_ * 14720, (long)B_ * (14720 + 3680),
                           (long)B_ * (14720 + 3680 + 920)};
  int col = h * DH_ + lane;
  float acc = 0.0f;
#pragma unroll
  for (int l = 0; l < 4; ++l) {
    int vw = vwA[l], vh = vhA[l];
    long lvlBase = baseRow[l] + (long)b * hwA[l];
    float invw = 1.0f / (float)vw, invh = 1.0f / (float)vh;
#pragma unroll
    for (int pt = 0; pt < 4; ++pt) {
      float ox = off[(long)bq * C_ + l * 64 + (h * 4 + pt) * 2 + 0];
      float oy = off[(long)bq * C_ + l * 64 + (h * 4 + pt) * 2 + 1];
      long ridx = ((long)((b * H_ + h) * 4 + pt) * 2) * Q_ + q;   // ref[256,2,1,900]
      float rx = ref[ridx];
      float ry = ref[ridx + Q_];
      float x = (rx + ox * invw) * (float)(vw - 1);   // (2*loc-1 +1)*0.5*(W-1)
      float y = (ry + oy * invh) * (float)(vh - 1);
      float x0f = floorf(x), y0f = floorf(y);
      float wx1 = x - x0f, wx0 = 1.0f - wx1;
      float wy1 = y - y0f, wy0 = 1.0f - wy1;
      int x0 = (int)x0f, y0 = (int)y0f;
      float w = aw[(long)bq * 128 + h * 16 + l * 4 + pt];   // sigmoid applied in GEMM
      float s = sample_corner(vproj, lvlBase, vw, vh, x0,     y0,     wx0 * wy0, col)
              + sample_corner(vproj, lvlBase, vw, vh, x0 + 1, y0,     wx1 * wy0, col)
              + sample_corner(vproj, lvlBase, vw, vh, x0,     y0 + 1, wx0 * wy1, col)
              + sample_corner(vproj, lvlBase, vw, vh, x0 + 1, y0 + 1, wx1 * wy1, col);
      acc += w * s;
    }
  }
  outTok[(long)bq * C_ + col] = (_Float16)acc;
}

// ---------------------------------------------------------------------------
// Host orchestration
// ---------------------------------------------------------------------------
static inline GemmP mkGemm(const _Float16* A, long lda, long sA,
                           const _Float16* W, long ldw, long sW,
                           const float* bias, void* out, long ldo, long sO,
                           int M, int N, int K) {
  GemmP p;
  p.A = A; p.lda = lda; p.strideA = sA;
  p.W = W; p.ldw = ldw; p.strideW = sW;
  p.bias = bias; p.out = out; p.ldo = ldo; p.strideO = sO;
  p.M = M; p.N = N; p.K = K;
  p.Mtiles = M / 32; p.NG = (N + 63) / 64;
  return p;
}

#define LAUNCH_GEMM(MODE, P, BATCH)                                        \
  do {                                                                     \
    GemmP _p = (P);                                                        \
    int _wv = _p.Mtiles * _p.NG;                                           \
    dim3 _g((unsigned)((_wv + 7) / 8), (unsigned)(BATCH));                 \
    gemm_wmma<MODE><<<_g, 256, 0, stream>>>(_p);                           \
  } while (0)

extern "C" void kernel_launch(void* const* d_in, const int* in_sizes, int n_in,
                              void* d_out, int out_size, void* d_ws, size_t ws_size,
                              hipStream_t stream) {
  const float* tgt   = (const float*)d_in[0];
  const float* qpos  = (const float*)d_in[1];
  const float* ref   = (const float*)d_in[2];
  const float* vsrc[4] = {(const float*)d_in[3], (const float*)d_in[4],
                          (const float*)d_in[5], (const float*)d_in[6]};
  const float* so_w = (const float*)d_in[7];   const float* so_b = (const float*)d_in[8];
  const float* vp_w = (const float*)d_in[9];   const float* vp_b = (const float*)d_in[10];
  const float* aw_w = (const float*)d_in[11];  const float* aw_b = (const float*)d_in[12];
  const float* op_w = (const float*)d_in[13];  const float* op_b = (const float*)d_in[14];
  const float* wq = (const float*)d_in[15];    const float* bq = (const float*)d_in[16];
  const float* wk = (const float*)d_in[17];    const float* bk = (const float*)d_in[18];
  const float* wv = (const float*)d_in[19];    const float* bv = (const float*)d_in[20];
  const float* wo = (const float*)d_in[21];    const float* bo = (const float*)d_in[22];
  const float* g1 = (const float*)d_in[23];    const float* b1 = (const float*)d_in[24];
  const float* g2 = (const float*)d_in[25];    const float* b2 = (const float*)d_in[26];
  const float* g3 = (const float*)d_in[27];    const float* b3 = (const float*)d_in[28];
  const float* l1w = (const float*)d_in[29];   const float* l1b = (const float*)d_in[30];
  const float* l2w = (const float*)d_in[31];   const float* l2b = (const float*)d_in[32];

  const int  hwA[4]   = {14720, 3680, 920, 240};
  const long rowOff[4] = {0, (long)B_ * 14720, (long)B_ * (14720 + 3680),
                          (long)B_ * (14720 + 3680 + 920)};
  const long totRows = (long)B_ * (14720 + 3680 + 920 + 240);   // 156480

  // workspace carve-out (256B aligned slices)
  size_t cur = 0;
  char* base = (char*)d_ws;
  auto alloc = [&](size_t bytes) -> char* {
    cur = (cur + 255) & ~(size_t)255;
    char* p = base + cur;
    cur += bytes;
    return p;
  };
  _Float16* Xqk  = (_Float16*)alloc((size_t)M_ * C_ * 2);
  _Float16* Xv   = (_Float16*)alloc((size_t)M_ * C_ * 2);
  float*    T0   = (float*)   alloc((size_t)M_ * C_ * 4);
  float*    QPt  = (float*)   alloc((size_t)M_ * C_ * 4);
  _Float16* wqh  = (_Float16*)alloc(65536 * 2);
  _Float16* wkh  = (_Float16*)alloc(65536 * 2);
  _Float16* wvh  = (_Float16*)alloc(65536 * 2);
  _Float16* woh  = (_Float16*)alloc(65536 * 2);
  _Float16* sowh = (_Float16*)alloc(65536 * 2);
  _Float16* vpwh = (_Float16*)alloc(262144 * 2);
  _Float16* awwh = (_Float16*)alloc(32768 * 2);
  _Float16* opwh = (_Float16*)alloc(65536 * 2);
  _Float16* l1wh = (_Float16*)alloc(262144 * 2);
  _Float16* l2wh = (_Float16*)alloc(262144 * 2);
  _Float16* qh   = (_Float16*)alloc((size_t)B_ * H_ * QP_ * DH_ * 2);
  _Float16* kh   = (_Float16*)alloc((size_t)B_ * H_ * QP_ * DH_ * 2);
  _Float16* vT   = (_Float16*)alloc((size_t)B_ * H_ * DH_ * KP_ * 2);
  _Float16* S    = (_Float16*)alloc((size_t)B_ * H_ * QP_ * KP_ * 2);   // ~110 MB
  _Float16* attnTok = (_Float16*)alloc((size_t)M_ * C_ * 2);
  float*    tgt2a   = (float*)   alloc((size_t)M_ * C_ * 4);
  float*    x1      = (float*)   alloc((size_t)M_ * C_ * 4);
  _Float16* xq      = (_Float16*)alloc((size_t)M_ * C_ * 2);
  float*    offb    = (float*)   alloc((size_t)M_ * C_ * 4);
  float*    awb     = (float*)   alloc((size_t)M_ * 128 * 4);
  _Float16* Vtok    = (_Float16*)alloc((size_t)totRows * C_ * 2);       // ~80 MB
  _Float16* Vproj   = (_Float16*)alloc((size_t)totRows * C_ * 2);       // ~80 MB
  _Float16* ca      = (_Float16*)alloc((size_t)M_ * C_ * 2);
  float*    tgt2b   = (float*)   alloc((size_t)M_ * C_ * 4);
  float*    x2      = (float*)   alloc((size_t)M_ * C_ * 4);
  _Float16* x2h     = (_Float16*)alloc((size_t)M_ * C_ * 2);
  _Float16* hid     = (_Float16*)alloc((size_t)M_ * 1024 * 2);
  float*    ffo     = (float*)   alloc((size_t)M_ * C_ * 4);
  if (cur > ws_size) return;   // insufficient scratch: bail out safely

  auto cvt = [&](const float* s, _Float16* d, int n) {
    f32_to_f16_kernel<<<(n + 255) / 256, 256, 0, stream>>>(s, d, n);
  };
  cvt(wq, wqh, 65536);   cvt(wk, wkh, 65536);  cvt(wv, wvh, 65536);  cvt(wo, woh, 65536);
  cvt(so_w, sowh, 65536); cvt(vp_w, vpwh, 262144); cvt(aw_w, awwh, 32768);
  cvt(op_w, opwh, 65536); cvt(l1w, l1wh, 262144);  cvt(l2w, l2wh, 262144);

  pack_tokens_kernel<<<(M_ * C_) / 256, 256, 0, stream>>>(tgt, qpos, Xqk, Xv, T0, QPt);
  for (int l = 0; l < 4; ++l) {
    long total = (long)B_ * hwA[l] * C_;
    pack_values_kernel<<<(unsigned)((total + 255) / 256), 256, 0, stream>>>(
        vsrc[l], Vtok + rowOff[l] * C_, hwA[l], total);
  }

  // ---- self attention -----------------------------------------------------
  LAUNCH_GEMM(4, mkGemm(Xqk, C_, 0, wqh, C_, 0, bq, qh, 0, 0, M_, C_, C_), 1);
  LAUNCH_GEMM(4, mkGemm(Xqk, C_, 0, wkh, C_, 0, bk, kh, 0, 0, M_, C_, C_), 1);
  LAUNCH_GEMM(5, mkGemm(Xv,  C_, 0, wvh, C_, 0, bv, vT, 0, 0, M_, C_, C_), 1);
  // scores: per (b,h): [QP_,32] x [QP_,32]^T -> S[QP_,KP_]
  LAUNCH_GEMM(6, mkGemm(qh, DH_, (long)QP_ * DH_, kh, DH_, (long)QP_ * DH_,
                        nullptr, S, KP_, (long)QP_ * KP_, QP_, QP_, DH_), B_ * H_);
  softmax_kernel<<<B_ * H_ * QP_, 256, 0, stream>>>(S);
  // O = P x V: per (b,h): A=S[QP_,KP_], W=vT[32,KP_] -> scatter to tokens
  LAUNCH_GEMM(7, mkGemm(S, KP_, (long)QP_ * KP_, vT, KP_, (long)DH_ * KP_,
                        nullptr, attnTok, 0, 0, QP_, DH_, KP_), B_ * H_);
  LAUNCH_GEMM(0, mkGemm(attnTok, C_, 0, woh, C_, 0, bo, tgt2a, C_, 0, M_, C_, C_), 1);
  ln_kernel<<<M_, 256, 0, stream>>>(T0, tgt2a, g2, b2, x1, xq, QPt, nullptr);

  // ---- deformable cross attention ----------------------------------------
  LAUNCH_GEMM(0, mkGemm(xq, C_, 0, sowh, C_, 0, so_b, offb, C_, 0, M_, C_, C_), 1);
  LAUNCH_GEMM(1, mkGemm(xq, C_, 0, awwh, C_, 0, aw_b, awb, 128, 0, M_, 128, C_), 1);
  for (int l = 0; l < 4; ++l) {
    int rows = B_ * hwA[l];
    LAUNCH_GEMM(2, mkGemm(Vtok + rowOff[l] * C_, C_, 0, vpwh + (long)l * 65536, C_, 0,
                          vp_b + l * C_, Vproj + rowOff[l] * C_, C_, 0, rows, C_, C_), 1);
  }
  deform_sample_kernel<<<M_ * H_ / 8, 256, 0, stream>>>(offb, awb, ref, Vproj, ca);
  LAUNCH_GEMM(0, mkGemm(ca, C_, 0, opwh, C_, 0, op_b, tgt2b, C_, 0, M_, C_, C_), 1);
  ln_kernel<<<M_, 256, 0, stream>>>(x1, tgt2b, g1, b1, x2, x2h, nullptr, nullptr);

  // ---- FFN ----------------------------------------------------------------
  LAUNCH_GEMM(3, mkGemm(x2h, C_, 0, l1wh, C_, 0, l1b, hid, 1024, 0, M_, 1024, C_), 1);
  LAUNCH_GEMM(0, mkGemm(hid, 1024, 0, l2wh, 1024, 0, l2b, ffo, C_, 0, M_, C_, 1024), 1);
  ln_kernel<<<M_, 256, 0, stream>>>(x2, ffo, g3, b3, nullptr, nullptr, nullptr, (float*)d_out);

  (void)in_sizes; (void)n_in; (void)out_size;
}
